// LISAGON_33414845563012
// MI455X (gfx1250) — matrix operand order repaired and checked by reading, not compiled, not run on hardware
//
#include <hip/hip_runtime.h>
#include <hip/hip_bf16.h>

// ---------- CDNA5 WMMA types ----------
typedef __attribute__((ext_vector_type(16))) _Float16 v16h;
typedef __attribute__((ext_vector_type(8)))  float    v8f;
typedef __attribute__((ext_vector_type(4)))  unsigned int v4u;
typedef __attribute__((ext_vector_type(8)))  int      v8i;
typedef __attribute__((ext_vector_type(4)))  int      v4i;

// Problem constants (match reference setup_inputs)
#define NLAT   8192
#define QPB    65536       // queries per batch element (Q)
#define HID    128
#define WAVES  2           // waves (16-query tiles) per block

// Padded LDS row strides (ushorts). 72 -> 36 words/row, 136 -> 68 words/row:
// (36*n) mod 64 and (68*n) mod 64 are distinct over n=0..15 => conflict-free
// b128 fragment loads. Both are multiples of 8 ushorts => 16B-aligned rows.
#define STR1   72          // W1^T rows: K=64 (61 used) + 8 pad
#define STR2   136         // W2^T rows: K=128 + 8 pad
#define STRH   136         // activation rows: 128 cols + 8 pad
#define WLDS   ((STR1 + STR2) * HID)   // 26624 ushorts
#define WBYTES ((size_t)WLDS * 2)      // 53248 bytes

#if __has_builtin(__builtin_amdgcn_tensor_load_to_lds) && __has_builtin(__builtin_amdgcn_s_wait_tensorcnt)
#define HAVE_TDM 1
#else
#define HAVE_TDM 0
#endif

// The amdgpu-toolchain (clang-23 / therock-10.0 headers) builtin takes 6 args
// (extra int32x8 group); ROCm 7.2 clang-22 takes 5. The TDM header ships only
// with the 6-arg toolchain, so key the arity off its presence.
#if __has_include(<hip/amd_detail/amd_gfx1250_TDM.h>)
#define TDM_SIX_ARGS 1
#else
#define TDM_SIX_ARGS 0
#endif

__device__ __forceinline__ unsigned short f2h_bits(float f) {
    union { _Float16 h; unsigned short u; } z;
    z.h = (_Float16)f;
    return z.u;
}

// Load a v16h fragment from LDS: 16B at p, 16B at p+off2 (off2 in ushorts)
__device__ __forceinline__ v16h frag_ld(const unsigned short* p, int off2) {
    union { v16h v; uint4 q[2]; } u;
    u.q[0] = *(const uint4*)(p);
    u.q[1] = *(const uint4*)(p + off2);
    return u.v;
}

// ---- prep kernel: transpose + f32->f16 + pad weights into d_ws (once) ----
// Layout: [ W1t : n*STR1 + k (k<61 valid) | W2t : n*STR2 + k (k<128 valid) ]
__global__ void liif_prep_weights(const float* __restrict__ W1,
                                  const float* __restrict__ W2,
                                  unsigned short* __restrict__ ws) {
    const int total = (STR1 + STR2) * HID;
    for (int e = blockIdx.x * blockDim.x + threadIdx.x; e < total;
         e += gridDim.x * blockDim.x) {
        unsigned short v;
        if (e < STR1 * HID) {
            int n = e / STR1, k = e % STR1;
            v = (k < 61) ? f2h_bits(W1[k * HID + n]) : (unsigned short)0;
        } else {
            int e2 = e - STR1 * HID;
            int n = e2 / STR2, k = e2 % STR2;
            v = (k < HID) ? f2h_bits(W2[k * HID + n]) : (unsigned short)0;
        }
        ws[e] = v;
    }
}

__global__ __launch_bounds__(32 * WAVES)
void liif_mlp_wmma(const float* __restrict__ coord,
                   const float* __restrict__ latent,
                   const float* __restrict__ W1, const float* __restrict__ b1,
                   const float* __restrict__ W2, const float* __restrict__ b2,
                   const float* __restrict__ W3, const float* __restrict__ b3,
                   float* __restrict__ out, int BQ,
                   const unsigned short* __restrict__ wsw, int useWS) {
    __shared__ __align__(16) unsigned short sWt[WLDS];            // 52 KB weights
    __shared__ __align__(16) unsigned short sH[WAVES][16 * STRH]; // 8.5 KB
    __shared__ float sB1[HID], sB2[HID], sW3[HID];
    __shared__ float sPred[WAVES][2][16];

    const int tid  = threadIdx.x;
    const int w    = tid >> 5;
    const int lane = tid & 31;
    const int half = lane >> 4;
    const int lrow = lane & 15;
    const int NT   = 32 * WAVES;

    // ---------------- stage weights into LDS ----------------
    if (useWS) {
#if HAVE_TDM
        if (tid < 32) {
            // Tensor Data Mover: one contiguous 53KB tile, global -> LDS.
            const unsigned lds_off = (unsigned)(size_t)(const void*)&sWt[0];
            const unsigned long long ga = (unsigned long long)(size_t)wsw;
            const unsigned e8 = (unsigned)(WBYTES >> 3);   // 8-byte elements

            v4u g0;
            g0[0] = 1u;                                    // count=1 (valid D#)
            g0[1] = lds_off;                               // lds_addr
            g0[2] = (unsigned)(ga & 0xFFFFFFFFu);          // global_addr lo
            g0[3] = (unsigned)((ga >> 32) & 0x1FFFFFFu)    // global_addr hi
                  | (2u << 30);                            // type=2 (image)
            v8i g1;
            g1[0] = (int)(3u << 16);                       // data_size=8B
            g1[1] = (int)((e8 & 0xFFFFu) << 16);           // tensor_dim0 lo16
            g1[2] = (int)((e8 >> 16) | (1u << 16));        // dim0 hi16 | tensor_dim1=1
            g1[3] = (int)(e8 << 16);                       // tile_dim0 = e8 (6656)
            g1[4] = 1;                                     // tile_dim1=1, tile_dim2=0
            g1[5] = (int)e8;                               // tensor_dim0_stride lo32
            g1[6] = 0;                                     // stride0 hi / stride1 lo
            g1[7] = 0;
            v4i gz4 = {0, 0, 0, 0};
#if TDM_SIX_ARGS
            v8i gz8 = {0, 0, 0, 0, 0, 0, 0, 0};
            __builtin_amdgcn_tensor_load_to_lds(g0, g1, gz4, gz4, gz8, 0);
#else
            __builtin_amdgcn_tensor_load_to_lds(g0, g1, gz4, gz4, 0);
#endif
            __builtin_amdgcn_s_wait_tensorcnt((short)0);
        }
#else
        // Coalesced vector copy fallback (f16 already packed in ws).
        for (int e = tid; e < WLDS / 8; e += NT)
            ((uint4*)sWt)[e] = ((const uint4*)wsw)[e];
#endif
    } else {
        // In-kernel f32 transpose/convert fallback (no usable workspace).
        for (int e = tid; e < STR1 * HID; e += NT) {
            int n = e / STR1, k = e % STR1;
            sWt[e] = (k < 61) ? f2h_bits(W1[k * HID + n]) : (unsigned short)0;
        }
        for (int e = tid; e < STR2 * HID; e += NT) {
            int n = e / STR2, k = e % STR2;
            sWt[STR1 * HID + e] = (k < HID) ? f2h_bits(W2[k * HID + n]) : (unsigned short)0;
        }
    }
    for (int i = tid; i < HID; i += NT) {
        sB1[i] = b1[i]; sB2[i] = b2[i]; sW3[i] = W3[i];
    }
    __syncthreads();

    const unsigned short* sW1t = sWt;
    const unsigned short* sW2t = sWt + STR1 * HID;
    unsigned short* hb = &sH[w][0];
    const float b3v = b3[0];
    const int ntiles = BQ >> 4;

    // -------------- persistent tile loop (amortizes staging) --------------
    for (int tile = blockIdx.x * WAVES + w; tile < ntiles;
         tile += gridDim.x * WAVES) {
        const int g    = tile * 16 + lrow;
        const int bidx = g / QPB;
        float area[2] = {0.f, 0.f};
        float c0 = 0.f;
        if (lane < 16) c0 = coord[g];

        #pragma unroll
        for (int s = 0; s < 2; ++s) {
            const float vx = (s == 0) ? -2.0f : 0.0f;

            // ---- build input rows (lanes 0-15, one query row each) ----
            if (lane < 16) {
                float cs = c0 + vx * (1.0f / NLAT) + 1e-6f;
                cs = fminf(fmaxf(cs, -1.0f + 1e-6f), 1.0f - 1e-6f);
                float fpos = (cs + 1.0f) * (0.5f * NLAT) - 0.5f;
                int idx = (int)ceilf(fpos);
                idx = min(max(idx, 0), NLAT - 1);
                float qc  = -1.0f + (2.0f * (float)idx + 1.0f) * (1.0f / NLAT);
                float rel = (c0 - qc) * (float)NLAT;
                area[s] = fabsf(rel) + 1e-9f;

                unsigned short* row = hb + lrow * STRH;
                row[0] = f2h_bits(rel);
                #pragma unroll
                for (int j = 0; j < 6; ++j) {
                    float fx = rel * (float)(1 << j);
                    row[1 + 2 * j] = f2h_bits(__sinf(fx));
                    row[2 + 2 * j] = f2h_bits(__cosf(fx));
                }
                int im = max(idx - 1, 0), ip = min(idx + 1, NLAT - 1);
                const float* base = latent + (size_t)bidx * NLAT * 16;
                const float* rows3[3] = { base + (size_t)im * 16,
                                          base + (size_t)idx * 16,
                                          base + (size_t)ip * 16 };
                #pragma unroll
                for (int rI = 0; rI < 3; ++rI) {
                    #pragma unroll
                    for (int vq = 0; vq < 4; ++vq) {
                        float4 f4 = ((const float4*)rows3[rI])[vq];
                        int o = 13 + rI * 16 + vq * 4;
                        row[o + 0] = f2h_bits(f4.x);
                        row[o + 1] = f2h_bits(f4.y);
                        row[o + 2] = f2h_bits(f4.z);
                        row[o + 3] = f2h_bits(f4.w);
                    }
                }
                row[61] = 0; row[62] = 0; row[63] = 0;
            }
            asm volatile("s_wait_dscnt 0" ::: "memory");  // wave-lockstep fence

            // ---- layer 1: [16x64] x [64x128] ----
            v16h a1[2];
            #pragma unroll
            for (int c = 0; c < 2; ++c)
                a1[c] = frag_ld(hb + lrow * STRH + c * 32 + half * 8, 16);

            v8f acc[8];
            #pragma unroll
            for (int t = 0; t < 8; ++t) { v8f z = {}; acc[t] = z; }
            #pragma unroll
            for (int t = 0; t < 8; ++t) {
                const int n = t * 16 + lrow;
                #pragma unroll
                for (int c = 0; c < 2; ++c) {
                    v16h bf = frag_ld(sW1t + n * STR1 + c * 32 + half * 8, 16);
                    acc[t] = __builtin_amdgcn_wmma_f32_16x16x32_f16(
                        false, a1[c], false, bf, (short)0, acc[t], false, false);
                }
            }

            // ---- bias + ReLU -> restage activations [16 q][128 n] f16 ----
            #pragma unroll
            for (int t = 0; t < 8; ++t) {
                const int n = t * 16 + lrow;
                const float bb = sB1[n];
                #pragma unroll
                for (int r = 0; r < 8; ++r) {
                    float h = fmaxf(acc[t][r] + bb, 0.0f);
                    hb[(r + 8 * half) * STRH + n] = f2h_bits(h);
                }
            }
            asm volatile("s_wait_dscnt 0" ::: "memory");

            // ---- layer 2: [16x128] x [128x128] ----
            v16h a2[4];
            #pragma unroll
            for (int c = 0; c < 4; ++c)
                a2[c] = frag_ld(hb + lrow * STRH + c * 32 + half * 8, 16);

            v8f acc2[8];
            #pragma unroll
            for (int t = 0; t < 8; ++t) { v8f z = {}; acc2[t] = z; }
            #pragma unroll
            for (int t = 0; t < 8; ++t) {
                const int n = t * 16 + lrow;
                #pragma unroll
                for (int c = 0; c < 4; ++c) {
                    v16h bf = frag_ld(sW2t + n * STR2 + c * 32 + half * 8, 16);
                    acc2[t] = __builtin_amdgcn_wmma_f32_16x16x32_f16(
                        false, a2[c], false, bf, (short)0, acc2[t], false, false);
                }
            }

            // ---- layer 3: ReLU dot with W3 + shfl butterfly reduce ----
            float p[8];
            #pragma unroll
            for (int r = 0; r < 8; ++r) p[r] = 0.0f;
            #pragma unroll
            for (int t = 0; t < 8; ++t) {
                const int n = t * 16 + lrow;
                const float bb = sB2[n], wv = sW3[n];
                #pragma unroll
                for (int r = 0; r < 8; ++r)
                    p[r] += fmaxf(acc2[t][r] + bb, 0.0f) * wv;
            }
            #pragma unroll
            for (int r = 0; r < 8; ++r) {
                #pragma unroll
                for (int m = 1; m < 16; m <<= 1)
                    p[r] += __shfl_xor(p[r], m, 32);
            }
            if (lrow == 0) {
                #pragma unroll
                for (int r = 0; r < 8; ++r)
                    sPred[w][s][r + 8 * half] = p[r] + b3v;
            }
        }

        asm volatile("s_wait_dscnt 0" ::: "memory");
        // ---- local ensemble: swap areas before weighting ----
        if (lane < 16) {
            float p0 = sPred[w][0][lrow];
            float p1 = sPred[w][1][lrow];
            float tot = area[0] + area[1];
            out[g] = (p0 * area[1] + p1 * area[0]) / tot;
        }
    }
}

extern "C" void kernel_launch(void* const* d_in, const int* in_sizes, int n_in,
                              void* d_out, int out_size, void* d_ws, size_t ws_size,
                              hipStream_t stream) {
    (void)n_in; (void)out_size;
    const float* coord  = (const float*)d_in[0];
    const float* latent = (const float*)d_in[1];
    const float* W1     = (const float*)d_in[2];
    const float* b1     = (const float*)d_in[3];
    const float* W2     = (const float*)d_in[4];
    const float* b2     = (const float*)d_in[5];
    const float* W3     = (const float*)d_in[6];
    const float* b3     = (const float*)d_in[7];
    float* out = (float*)d_out;

    const int BQ     = in_sizes[0];          // B*Q = 262144
    const int ntiles = BQ / 16;
    const int useWS  = (d_ws != nullptr && ws_size >= WBYTES) ? 1 : 0;

    if (useWS) {
        liif_prep_weights<<<64, 256, 0, stream>>>(W1, W2, (unsigned short*)d_ws);
    }

    int blocks = ntiles / (WAVES * 8);       // ~8 tiles per wave (persistent)
    if (blocks < 1) blocks = 1;
    liif_mlp_wmma<<<blocks, 32 * WAVES, 0, stream>>>(
        coord, latent, W1, b1, W2, b2, W3, b3, out, BQ,
        (const unsigned short*)d_ws, useWS);
}
